// SelfAttention_83846351553176
// MI455X (gfx1250) — matrix operand order repaired and checked by reading, not compiled, not run on hardware
//
#include <hip/hip_runtime.h>
#include <hip/hip_bf16.h>

// ---------------- WMMA fragment helpers (gfx1250, wave32) ----------------
typedef __bf16 v16bf __attribute__((ext_vector_type(16)));
typedef float  v8f   __attribute__((ext_vector_type(8)));
typedef int    i32x4 __attribute__((ext_vector_type(4)));

// A-matrix 16x32 bf16: lane l holds row r=l&15; element j maps to
// K = (j>=8 ? 16 : 0) + (l>>4)*8 + (j&7)   (ISA 7.12.2, 16-bit A 16x32)
__device__ __forceinline__ v16bf loadA(const __bf16* base, int rs, int lane) {
  const int r = lane & 15, h = lane >> 4;
  v16bf f;
#pragma unroll
  for (int j = 0; j < 16; ++j) {
    const int k = ((j >> 3) << 4) + h * 8 + (j & 7);
    f[j] = base[r * rs + k];
  }
  return f;
}

// B-matrix 32x16 bf16: lane l holds output column r=l&15; element j maps to
// K = (j>=8 ? 16 : 0) + (l>>4)*8 + (j&7). addr = r*rs + k*ks.
__device__ __forceinline__ v16bf loadB(const __bf16* base, int rs, int ks, int lane) {
  const int r = lane & 15, h = lane >> 4;
  v16bf f;
#pragma unroll
  for (int j = 0; j < 16; ++j) {
    const int k = ((j >> 3) << 4) + h * 8 + (j & 7);
    f[j] = base[r * rs + k * ks];
  }
  return f;
}

__device__ __forceinline__ v8f wmma_bf16(v16bf a, v16bf b, v8f c) {
  return __builtin_amdgcn_wmma_f32_16x16x32_bf16(
      false, a, false, b, (short)0, c, false, false);
}

// ---------------- CDNA5 async global->LDS copy (ASYNCcnt path) -----------
#if defined(__gfx1250__) &&                                           \
    __has_builtin(__builtin_amdgcn_global_load_async_to_lds_b128) &&  \
    __has_builtin(__builtin_amdgcn_s_wait_asynccnt)
#define HAVE_ASYNC 1
#else
#define HAVE_ASYNC 0
#endif

// copy 16 bytes (8 bf16) per lane, global -> LDS, without touching VGPRs
__device__ __forceinline__ void cp_async16(__bf16* ldst, const __bf16* gsrc) {
#if HAVE_ASYNC
  __builtin_amdgcn_global_load_async_to_lds_b128(
      (__attribute__((address_space(1))) i32x4*)gsrc,
      (__attribute__((address_space(3))) i32x4*)ldst, 0, 0);
#else
  *(int4*)ldst = *(const int4*)gsrc;
#endif
}
__device__ __forceinline__ void wait_async_hi() {  // allow 6 outstanding (prefetch slab)
#if HAVE_ASYNC
  __builtin_amdgcn_s_wait_asynccnt(6);
#endif
}
__device__ __forceinline__ void wait_async_0() {
#if HAVE_ASYNC
  __builtin_amdgcn_s_wait_asynccnt(0);
#endif
}

// ---------------- Kernel 0: fp32 -> bf16 conversion ----------------------
__global__ void to_bf16(const float4* __restrict__ src, __bf16* __restrict__ dst,
                        int n4) {
  const int i = blockIdx.x * 256 + threadIdx.x;
  if (i < n4) {
    const float4 v = src[i];
    dst[i * 4 + 0] = (__bf16)v.x;
    dst[i * 4 + 1] = (__bf16)v.y;
    dst[i * 4 + 2] = (__bf16)v.z;
    dst[i * 4 + 3] = (__bf16)v.w;
  }
}

// ---------------- Shared GEMM core:  C[64x128] = X[64xK] * W[128xK]^T ----
// bf16 inputs, double-buffered LDS slabs filled by async copies.
// Block = 256 threads (8 waves). Wave w -> 32x32 sub-tile (wm=w>>2, wn=w&3).
// Per wave per slab: 2 (Xs) + 4 (Ws) async instructions -> wait threshold 6.
__device__ __forceinline__ void gemm_core(const __bf16* __restrict__ X,
                                          const __bf16* __restrict__ W,
                                          __bf16* Xs, __bf16* Ws,
                                          int m0, int n0, v8f c[2][2]) {
  const int tid = threadIdx.x, lane = tid & 31, w = tid >> 5;
  const int wm = w >> 2, wn = w & 3;

  auto issue = [&](int buf, int kt) {
    __bf16* xs = Xs + buf * (64 * 64);
    __bf16* ws = Ws + buf * (128 * 64);
#pragma unroll
    for (int i = 0; i < 2; ++i) {                 // 64x64 slab: 512 chunks
      const int ch = i * 256 + tid;
      const int r = ch >> 3, c8 = ch & 7;
      cp_async16(xs + ch * 8, X + (size_t)(m0 + r) * 1024 + kt + c8 * 8);
    }
#pragma unroll
    for (int i = 0; i < 4; ++i) {                 // 128x64 slab: 1024 chunks
      const int ch = i * 256 + tid;
      const int r = ch >> 3, c8 = ch & 7;
      cp_async16(ws + ch * 8, W + (size_t)(n0 + r) * 1024 + kt + c8 * 8);
    }
  };

  issue(0, 0);
  for (int it = 0; it < 16; ++it) {
    const int cur = it & 1;
    if (it + 1 < 16) { issue(cur ^ 1, (it + 1) * 64); wait_async_hi(); }
    else             { wait_async_0(); }
    __syncthreads();                               // slab `cur` visible to all
    const __bf16* xs = Xs + cur * (64 * 64);
    const __bf16* ws = Ws + cur * (128 * 64);
#pragma unroll
    for (int ks = 0; ks < 64; ks += 32) {
      const v16bf a0 = loadA(xs + (wm * 32 +  0) * 64 + ks, 64, lane);
      const v16bf a1 = loadA(xs + (wm * 32 + 16) * 64 + ks, 64, lane);
      const v16bf b0 = loadB(ws + (wn * 32 +  0) * 64 + ks, 64, 1, lane);
      const v16bf b1 = loadB(ws + (wn * 32 + 16) * 64 + ks, 64, 1, lane);
      c[0][0] = wmma_bf16(a0, b0, c[0][0]);
      c[0][1] = wmma_bf16(a0, b1, c[0][1]);
      c[1][0] = wmma_bf16(a1, b0, c[1][0]);
      c[1][1] = wmma_bf16(a1, b1, c[1][1]);
    }
    __syncthreads();                               // reads done -> slab reusable
  }
}

// ---------------- Kernel 1: fused QKV projection -------------------------
// grid (64, 8, 3): z selects Wq/Wk/Wv. Q,K -> fp32 (LN input), V -> bf16.
__global__ void gemm_qkv(const __bf16* __restrict__ xb,
                         const __bf16* __restrict__ Wqb,
                         const __bf16* __restrict__ Wkb,
                         const __bf16* __restrict__ Wvb,
                         float* __restrict__ qf, float* __restrict__ kf,
                         __bf16* __restrict__ vb) {
  __shared__ __align__(16) __bf16 Xs[2 * 64 * 64];
  __shared__ __align__(16) __bf16 Ws[2 * 128 * 64];
  const int z = blockIdx.z;
  const __bf16* W = (z == 0) ? Wqb : (z == 1) ? Wkb : Wvb;
  const int m0 = blockIdx.x * 64, n0 = blockIdx.y * 128;
  v8f c[2][2] = {{{}, {}}, {{}, {}}};
  gemm_core(xb, W, Xs, Ws, m0, n0, c);
  const int lane = threadIdx.x & 31, w = threadIdx.x >> 5;
  const int wm = w >> 2, wn = w & 3;
#pragma unroll
  for (int ti = 0; ti < 2; ++ti)
#pragma unroll
    for (int tj = 0; tj < 2; ++tj)
#pragma unroll
      for (int i = 0; i < 8; ++i) {
        const int m = m0 + wm * 32 + ti * 16 + (lane >> 4) * 8 + i;
        const int n = n0 + wn * 32 + tj * 16 + (lane & 15);
        const int b = m >> 11, s = m & 2047, hh = n >> 6, dd = n & 63;
        const size_t o = (((size_t)(b * 16 + hh)) * 2048 + s) * 64 + dd;
        if (z == 0)      qf[o] = c[ti][tj][i];
        else if (z == 1) kf[o] = c[ti][tj][i];
        else             vb[o] = (__bf16)c[ti][tj][i];
      }
}

// ---------------- Kernel 2: per-head LayerNorm + RoPE on Q and K ---------
// One wave per 64-element row (fp32 in, bf16 out).
__global__ void ln_rope(const float* __restrict__ qf, const float* __restrict__ kf,
                        __bf16* __restrict__ qb, __bf16* __restrict__ kb,
                        const float* __restrict__ qg, const float* __restrict__ qbeta,
                        const float* __restrict__ kg, const float* __restrict__ kbeta) {
  const int wid  = blockIdx.x * 8 + (threadIdx.x >> 5);
  const int lane = threadIdx.x & 31;
  const int t = wid >> 16;                 // 0 -> Q, 1 -> K (65536 rows each)
  const int r = wid & 65535;
  const float* p  = (t ? kf : qf) + (size_t)r * 64;
  __bf16* po      = (t ? kb : qb) + (size_t)r * 64;
  const float* g  = t ? kg : qg;
  const float* be = t ? kbeta : qbeta;
  const int s = r & 2047;

  float y0 = p[lane], y1 = p[lane + 32];
  float sum = y0 + y1, sq = y0 * y0 + y1 * y1;
#pragma unroll
  for (int m = 16; m >= 1; m >>= 1) {
    sum += __shfl_xor(sum, m, 32);
    sq  += __shfl_xor(sq,  m, 32);
  }
  const float mu   = sum * (1.0f / 64.0f);
  const float var  = sq * (1.0f / 64.0f) - mu * mu;
  const float rstd = rsqrtf(var + 1e-5f);
  y0 = (y0 - mu) * rstd * g[lane]      + be[lane];
  y1 = (y1 - mu) * rstd * g[lane + 32] + be[lane + 32];

  // rotate_half: rot[j] = (j < 32) ? -y[2j+1] : y[2(j-32)]
  const int j1 = 2 * lane + 1;
  const float a0 = __shfl(y0, j1 & 31, 32);
  const float a1 = __shfl(y1, j1 & 31, 32);
  const float rot0 = -((j1 < 32) ? a0 : a1);
  const int j2 = 2 * lane;
  const float b0 = __shfl(y0, j2 & 31, 32);
  const float b1 = __shfl(y1, j2 & 31, 32);
  const float rot1 = (j2 < 32) ? b0 : b1;

  const float pos = (float)s;
  const float L = 9.210340371976184f;      // ln(10000)
  const float th0 = pos * expf(-((float)(lane >> 1))        * (L / 32.0f));
  const float th1 = pos * expf(-((float)(16 + (lane >> 1))) * (L / 32.0f));
  po[lane]      = (__bf16)(y0 * cosf(th0) + rot0 * sinf(th0));
  po[lane + 32] = (__bf16)(y1 * cosf(th1) + rot1 * sinf(th1));
}

// ---------------- Kernel 3: flash attention (causal) ---------------------
// grid (32, 32); block 256 = 8 waves. bf16 Q/K/V staged via async copies.
__global__ void attn(const __bf16* __restrict__ qb, const __bf16* __restrict__ kb,
                     const __bf16* __restrict__ vb, __bf16* __restrict__ att) {
  __shared__ __align__(16) __bf16 Qs[64 * 64];
  __shared__ __align__(16) __bf16 Ks[64 * 64];
  __shared__ __align__(16) __bf16 Vs[64 * 64];
  __shared__ __align__(16) __bf16 Ps[64 * 64];
  __shared__ float Ss[64 * 64];
  __shared__ float mrow[64], lrow[64], arow[64];

  const int tid = threadIdx.x, lane = tid & 31, w = tid >> 5;
  const int wm = w >> 1, wn = w & 1;
  const int bh = blockIdx.y, qt = blockIdx.x, q0 = qt * 64;
  const size_t base = (size_t)bh * 2048 * 64;
  const float NEG = -__builtin_inff();

#pragma unroll
  for (int i = 0; i < 2; ++i) {                      // Q tile: 512 chunks
    const int ch = i * 256 + tid;
    const int r = ch >> 3, c8 = ch & 7;
    cp_async16(Qs + ch * 8, qb + base + (size_t)(q0 + r) * 64 + c8 * 8);
  }
  if (tid < 64) { mrow[tid] = NEG; lrow[tid] = 0.0f; }

  v8f o0 = {}, o1 = {};
  for (int j = 0; j <= qt; ++j) {
    const int k0 = j * 64;
    __syncthreads();                                 // protect Ks/Vs/Ps
#pragma unroll
    for (int i = 0; i < 2; ++i) {
      const int ch = i * 256 + tid;
      const int r = ch >> 3, c8 = ch & 7;
      cp_async16(Ks + ch * 8, kb + base + (size_t)(k0 + r) * 64 + c8 * 8);
      cp_async16(Vs + ch * 8, vb + base + (size_t)(k0 + r) * 64 + c8 * 8);
    }
    wait_async_0();
    __syncthreads();

    // S = Q K^T
    v8f s0 = {}, s1 = {};
#pragma unroll
    for (int ks = 0; ks < 64; ks += 32) {
      const v16bf a  = loadA(Qs + (wm * 16) * 64 + ks, 64, lane);
      const v16bf b0 = loadB(Ks + (wn * 32 +  0) * 64 + ks, 64, 1, lane);
      const v16bf b1 = loadB(Ks + (wn * 32 + 16) * 64 + ks, 64, 1, lane);
      s0 = wmma_bf16(a, b0, s0);
      s1 = wmma_bf16(a, b1, s1);
    }
    // scale + causal mask -> LDS
    {
      const int rb = wm * 16 + (lane >> 4) * 8;
      const int c0 = wn * 32 + (lane & 15);
#pragma unroll
      for (int i = 0; i < 8; ++i) {
        const int row = rb + i, gq = q0 + row;
        Ss[row * 64 + c0]      = (k0 + c0      <= gq) ? s0[i] * 0.125f : NEG;
        Ss[row * 64 + c0 + 16] = (k0 + c0 + 16 <= gq) ? s1[i] * 0.125f : NEG;
      }
    }
    __syncthreads();

    // online softmax: 4 threads per row, 16 cols each
    {
      const int row = tid >> 2, cs = (tid & 3) * 16;
      float mx = NEG;
#pragma unroll
      for (int cc = 0; cc < 16; ++cc) mx = fmaxf(mx, Ss[row * 64 + cs + cc]);
      mx = fmaxf(mx, __shfl_xor(mx, 1, 32));
      mx = fmaxf(mx, __shfl_xor(mx, 2, 32));
      const float mo = mrow[row];
      const float mn = fmaxf(mo, mx);
      const float al = __expf(mo - mn);
      float psum = 0.0f;
#pragma unroll
      for (int cc = 0; cc < 16; ++cc) {
        const float pv = __expf(Ss[row * 64 + cs + cc] - mn);
        Ps[row * 64 + cs + cc] = (__bf16)pv;
        psum += pv;
      }
      psum += __shfl_xor(psum, 1, 32);
      psum += __shfl_xor(psum, 2, 32);
      if ((tid & 3) == 0) {
        mrow[row] = mn;
        arow[row] = al;
        lrow[row] = lrow[row] * al + psum;
      }
    }
    __syncthreads();

    // rescale O, then O += P V  (V read column-wise: addr = k*64 + d)
    {
      const int rb = wm * 16 + (lane >> 4) * 8;
#pragma unroll
      for (int i = 0; i < 8; ++i) {
        const float al = arow[rb + i];
        o0[i] *= al;
        o1[i] *= al;
      }
    }
#pragma unroll
    for (int ks = 0; ks < 64; ks += 32) {
      const v16bf a  = loadA(Ps + (wm * 16) * 64 + ks, 64, lane);
      const v16bf b0 = loadB(Vs + ks * 64 + wn * 32,      1, 64, lane);
      const v16bf b1 = loadB(Vs + ks * 64 + wn * 32 + 16, 1, 64, lane);
      o0 = wmma_bf16(a, b0, o0);
      o1 = wmma_bf16(a, b1, o1);
    }
  }
  __syncthreads();

  // epilogue: O /= l ; write bf16 att in (B*S, D) layout
  const int b = bh >> 4, hh = bh & 15;
  const int rb = wm * 16 + (lane >> 4) * 8;
  const int cc = lane & 15;
#pragma unroll
  for (int i = 0; i < 8; ++i) {
    const int row = rb + i;
    const float inv = 1.0f / lrow[row];
    const int s = q0 + row;
    const size_t ob = ((size_t)(b * 2048 + s)) * 1024 + hh * 64;
    att[ob + wn * 32 + cc]      = (__bf16)(o0[i] * inv);
    att[ob + wn * 32 + 16 + cc] = (__bf16)(o1[i] * inv);
  }
}

// ---------------- Kernel 4: output projection + bias ---------------------
__global__ void gemm_out(const __bf16* __restrict__ attv,
                         const __bf16* __restrict__ Wob,
                         const float* __restrict__ bo,
                         float* __restrict__ out) {
  __shared__ __align__(16) __bf16 Xs[2 * 64 * 64];
  __shared__ __align__(16) __bf16 Ws[2 * 128 * 64];
  const int m0 = blockIdx.x * 64, n0 = blockIdx.y * 128;
  v8f c[2][2] = {{{}, {}}, {{}, {}}};
  gemm_core(attv, Wob, Xs, Ws, m0, n0, c);
  const int lane = threadIdx.x & 31, w = threadIdx.x >> 5;
  const int wm = w >> 2, wn = w & 3;
#pragma unroll
  for (int ti = 0; ti < 2; ++ti)
#pragma unroll
    for (int tj = 0; tj < 2; ++tj)
#pragma unroll
      for (int i = 0; i < 8; ++i) {
        const int m = m0 + wm * 32 + ti * 16 + (lane >> 4) * 8 + i;
        const int n = n0 + wn * 32 + tj * 16 + (lane & 15);
        out[(size_t)m * 1024 + n] = c[ti][tj][i] + bo[n];
      }
}

// ---------------- Host-side launch ---------------------------------------
extern "C" void kernel_launch(void* const* d_in, const int* in_sizes, int n_in,
                              void* d_out, int out_size, void* d_ws, size_t ws_size,
                              hipStream_t stream) {
  (void)in_sizes; (void)n_in; (void)out_size; (void)ws_size;
  const float* x  = (const float*)d_in[0];
  const float* Wq = (const float*)d_in[1];
  const float* Wk = (const float*)d_in[2];
  const float* Wv = (const float*)d_in[3];
  const float* qg = (const float*)d_in[4];
  const float* qb_ = (const float*)d_in[5];
  const float* kg = (const float*)d_in[6];
  const float* kb_ = (const float*)d_in[7];
  const float* Wo = (const float*)d_in[8];
  const float* bo = (const float*)d_in[9];
  float* out = (float*)d_out;

  // workspace layout (all offsets MB-aligned)
  char* p = (char*)d_ws;
  __bf16* xb  = (__bf16*)p; p += (size_t)4096 * 1024 * 2;   // 8 MB
  __bf16* wqb = (__bf16*)p; p += (size_t)1024 * 1024 * 2;   // 2 MB
  __bf16* wkb = (__bf16*)p; p += (size_t)1024 * 1024 * 2;
  __bf16* wvb = (__bf16*)p; p += (size_t)1024 * 1024 * 2;
  __bf16* wob = (__bf16*)p; p += (size_t)1024 * 1024 * 2;
  float*  qf  = (float*)p;  p += (size_t)4096 * 1024 * 4;   // 16 MB
  float*  kf  = (float*)p;  p += (size_t)4096 * 1024 * 4;   // 16 MB
  __bf16* qbv = (__bf16*)p; p += (size_t)4096 * 1024 * 2;   // 8 MB
  __bf16* kbv = (__bf16*)p; p += (size_t)4096 * 1024 * 2;
  __bf16* vbv = (__bf16*)p; p += (size_t)4096 * 1024 * 2;
  __bf16* attb = (__bf16*)qf;  // qf dead after ln_rope; alias for att buffer

  // 0) one-time bf16 conversion of activations + weights
  to_bf16<<<dim3(4096), 256, 0, stream>>>((const float4*)x,  xb,  1024 * 1024);
  to_bf16<<<dim3(1024), 256, 0, stream>>>((const float4*)Wq, wqb, 256 * 1024);
  to_bf16<<<dim3(1024), 256, 0, stream>>>((const float4*)Wk, wkb, 256 * 1024);
  to_bf16<<<dim3(1024), 256, 0, stream>>>((const float4*)Wv, wvb, 256 * 1024);
  to_bf16<<<dim3(1024), 256, 0, stream>>>((const float4*)Wo, wob, 256 * 1024);

  gemm_qkv<<<dim3(64, 8, 3), 256, 0, stream>>>(xb, wqb, wkb, wvb, qf, kf, vbv);
  ln_rope <<<dim3(16384),    256, 0, stream>>>(qf, kf, qbv, kbv, qg, qb_, kg, kb_);
  attn    <<<dim3(32, 32),   256, 0, stream>>>(qbv, kbv, vbv, attb);
  gemm_out<<<dim3(64, 8),    256, 0, stream>>>(attb, wob, bo, out);
}
